// PhysicsMessagePassing_66554813219005
// MI455X (gfx1250) — compile-verified, matrix operand order checked by reading, not run on hardware
//
#include <hip/hip_runtime.h>

#define NN 100000
#define NE 1600000

typedef unsigned short u16;
typedef unsigned int   u32;
typedef __attribute__((ext_vector_type(16))) __bf16 v16bf;
typedef __attribute__((ext_vector_type(2)))  __bf16 v2bf;
typedef __attribute__((ext_vector_type(8)))  float  v8f;

struct BF32 { uint4 lo; uint4 hi; };   // 32B carrier for a v16bf fragment

// ---------- scalar helpers (native bf16 conversion; backend emits v_cvt_*) ----------
__device__ __forceinline__ u16 f2bf(float f) {
  return __builtin_bit_cast(u16, (__bf16)f);
}
__device__ __forceinline__ u32 pk(float a, float b) {
  v2bf t; t[0] = (__bf16)a; t[1] = (__bf16)b;
  return __builtin_bit_cast(u32, t);
}
// x * sigmoid(x) with v_exp_f32 (base-2) + v_rcp_f32 trans ops
__device__ __forceinline__ float silu(float v) {
  float e = __builtin_amdgcn_exp2f(v * -1.442695041f);
  return v * __builtin_amdgcn_rcpf(1.0f + e);
}

// A fragment: pre-swizzled weights, 16 contiguous halves per lane per tile
__device__ __forceinline__ v16bf ld_fragA(const u16* swz, int tile, int lane) {
  const u16* p = swz + (((tile << 5) + lane) << 4);
  BF32 s; s.lo = *(const uint4*)p; s.hi = *(const uint4*)(p + 8);
  return __builtin_bit_cast(v16bf, s);
}
// B fragment from a bf16 activation row in LDS: chunks [k0+half*8 .. +7] and +16
__device__ __forceinline__ v16bf ld_fragB(const u16* row, int k0, int half) {
  const u16* p = row + k0 + half * 8;
  BF32 s; s.lo = *(const uint4*)p; s.hi = *(const uint4*)(p + 16);
  return __builtin_bit_cast(v16bf, s);
}
// B fragment built directly from a 32-float global row (channel 31 replaced by v31)
__device__ __forceinline__ v16bf mk_fragB_g(const float* __restrict__ row, int half, float v31) {
  const float4* p = (const float4*)(row + half * 8);
  float4 a0 = p[0], a1 = p[1];
  const float4* q = (const float4*)(row + 16 + half * 8);
  float4 b0 = q[0], b1 = q[1];
  b1.w = half ? v31 : b1.w;                 // channel 31 lives in the half==1 chunk
  BF32 s;
  s.lo = make_uint4(pk(a0.x, a0.y), pk(a0.z, a0.w), pk(a1.x, a1.y), pk(a1.z, a1.w));
  s.hi = make_uint4(pk(b0.x, b0.y), pk(b0.z, b0.w), pk(b1.x, b1.y), pk(b1.z, b1.w));
  return __builtin_bit_cast(v16bf, s);
}
// acc initialized with per-output-channel bias (rows mt*16 + 8*half + j)
__device__ __forceinline__ v8f bias_acc(const float* __restrict__ b, int mt, int half) {
  const float4* p = (const float4*)(b + mt * 16 + half * 8);
  float4 u = p[0], v = p[1];
  v8f a = {u.x, u.y, u.z, u.w, v.x, v.y, v.z, v.w};
  return a;
}
// SiLU + pack 8 outputs + single b128 LDS store
__device__ __forceinline__ void silu_store(u16* dstp, v8f acc) {
  uint4 qo = make_uint4(pk(silu(acc[0]), silu(acc[1])), pk(silu(acc[2]), silu(acc[3])),
                        pk(silu(acc[4]), silu(acc[5])), pk(silu(acc[6]), silu(acc[7])));
  *(uint4*)dstp = qo;
}

#define HS 72   // padded LDS activation row stride (halves): 144B, distinct bank groups

// ---------- utility kernels ----------
__global__ void k_zero(float* __restrict__ p, int n) {
  int i = blockIdx.x * blockDim.x + threadIdx.x;
  if (i < n) p[i] = 0.0f;
}
__global__ void k_take31(const float* __restrict__ s, float* __restrict__ d, int n) {
  int i = blockIdx.x * blockDim.x + threadIdx.x;
  if (i < n) d[i] = s[(size_t)i * 32 + 31];
}
__global__ void k_aggconst(const float* __restrict__ ea, const int* __restrict__ dst,
                           float* __restrict__ aggc) {
  int e = blockIdx.x * blockDim.x + threadIdx.x;
  if (e >= NE) return;
  int d = dst[e];
  const float* row = ea + (size_t)e * 32;
#pragma unroll
  for (int c = 0; c < 31; ++c) atomicAdd(&aggc[(size_t)d * 32 + c], row[c]);
}
// swizzle W[K][N] (row-major, K mult of 32, N mult of 16) into A-fragment order:
// tile t = mt*(K/32)+kt ; lane L=(m,half) ; 16 halves contiguous
__global__ void k_swz(const float* __restrict__ W, u16* __restrict__ out, int K, int N) {
  int idx = blockIdx.x * blockDim.x + threadIdx.x;
  if (idx >= K * N) return;
  int i = idx & 15, L = (idx >> 4) & 31, t = idx >> 9;
  int nkt = K >> 5;
  int kt = t % nkt, mt = t / nkt;
  int m = L & 15, half = L >> 4;
  int k = kt * 32 + ((i < 8) ? (half * 8 + i) : (16 + half * 8 + (i - 8)));
  out[idx] = f2bf(W[(size_t)k * N + mt * 16 + m]);
}
// swizzle a [K][1] output-layer weight as a 16-wide tile with cols 1..15 zero
__global__ void k_swz_vec(const float* __restrict__ w, u16* __restrict__ out, int K) {
  int idx = blockIdx.x * blockDim.x + threadIdx.x;
  if (idx >= K * 16) return;
  int i = idx & 15, L = (idx >> 4) & 31, kt = idx >> 9;
  int m = L & 15, half = L >> 4;
  int k = kt * 32 + ((i < 8) ? (half * 8 + i) : (16 + half * 8 + (i - 8)));
  out[idx] = (m == 0) ? f2bf(w[k]) : (u16)0;
}

// ---------- edge processor: [x_s|x_r|ea](96) -> 64 -> 64 -> 1 ----------
__global__ __launch_bounds__(128, 1) void edge_kernel(
    const float* __restrict__ x, const float* __restrict__ ea,
    const int* __restrict__ src, const int* __restrict__ dst,
    const float* __restrict__ x31, float* __restrict__ e31,
    float* __restrict__ agg31,
    const u16* __restrict__ W1s, const float* __restrict__ b1,
    const u16* __restrict__ W2s, const float* __restrict__ b2,
    const u16* __restrict__ W3s, const float* __restrict__ b3) {
  __shared__ __align__(16) u16 sW1[96 * 64];
  __shared__ __align__(16) u16 sW2[64 * 64];
  __shared__ __align__(16) u16 sW3[16 * 64];
  __shared__ __align__(16) u16 sH1[4][16 * HS];
  __shared__ __align__(16) u16 sH2[4][16 * HS];

  const int tid = threadIdx.x;
  const int wave = tid >> 5, lane = tid & 31;
  const int m = lane & 15, half = lane >> 4;   // m = item (edge) index / D column

  { // stage pre-swizzled weights (already in load order -> straight uint4 copies)
    uint4* s1 = (uint4*)sW1; const uint4* g1 = (const uint4*)W1s;
    for (int i = tid; i < (96 * 64) / 8; i += 128) s1[i] = g1[i];
    uint4* s2 = (uint4*)sW2; const uint4* g2 = (const uint4*)W2s;
    for (int i = tid; i < (64 * 64) / 8; i += 128) s2[i] = g2[i];
    uint4* s3 = (uint4*)sW3; const uint4* g3 = (const uint4*)W3s;
    for (int i = tid; i < (16 * 64) / 8; i += 128) s3[i] = g3[i];
  }

  const int ebase = blockIdx.x * 64 + wave * 16;   // NE == 25000*64: always in range
  const int e = ebase + m;
  const int s = src[e];
  const int d = dst[e];
  const float e31v = e31[e];

  // layer-1 B fragments straight from global (no feat staging needed)
  v16bf bxs = mk_fragB_g(x + (size_t)s * 32, half, x31[s]);
  v16bf bxr = mk_fragB_g(x + (size_t)d * 32, half, x31[d]);
  v16bf bed = mk_fragB_g(ea + (size_t)e * 32, half, e31v);

  u16* h1 = sH1[wave];
  u16* h2 = sH2[wave];
  __syncthreads();   // weights staged

  // layer 1: 96 -> 64  (A = swizzled W1^T tiles, B = activations)
#pragma unroll
  for (int mt = 0; mt < 4; ++mt) {
    v8f acc = bias_acc(b1, mt, half);
    acc = __builtin_amdgcn_wmma_f32_16x16x32_bf16(false, ld_fragA(sW1, mt * 3 + 0, lane), false, bxs, (short)0, acc, false, false);
    acc = __builtin_amdgcn_wmma_f32_16x16x32_bf16(false, ld_fragA(sW1, mt * 3 + 1, lane), false, bxr, (short)0, acc, false, false);
    acc = __builtin_amdgcn_wmma_f32_16x16x32_bf16(false, ld_fragA(sW1, mt * 3 + 2, lane), false, bed, (short)0, acc, false, false);
    silu_store(h1 + m * HS + mt * 16 + half * 8, acc);
  }
  __syncthreads();

  // layer 2: 64 -> 64
  {
    v16bf c0 = ld_fragB(h1 + m * HS, 0, half);
    v16bf c1 = ld_fragB(h1 + m * HS, 32, half);
#pragma unroll
    for (int mt = 0; mt < 4; ++mt) {
      v8f acc = bias_acc(b2, mt, half);
      acc = __builtin_amdgcn_wmma_f32_16x16x32_bf16(false, ld_fragA(sW2, mt * 2 + 0, lane), false, c0, (short)0, acc, false, false);
      acc = __builtin_amdgcn_wmma_f32_16x16x32_bf16(false, ld_fragA(sW2, mt * 2 + 1, lane), false, c1, (short)0, acc, false, false);
      silu_store(h2 + m * HS + mt * 16 + half * 8, acc);
    }
  }
  __syncthreads();

  // layer 3: 64 -> 1 via zero-padded W3 tile; result in acc[0] of lanes 0..15
  {
    v16bf d0 = ld_fragB(h2 + m * HS, 0, half);
    v16bf d1 = ld_fragB(h2 + m * HS, 32, half);
    v8f a3 = {};
    a3 = __builtin_amdgcn_wmma_f32_16x16x32_bf16(false, ld_fragA(sW3, 0, lane), false, d0, (short)0, a3, false, false);
    a3 = __builtin_amdgcn_wmma_f32_16x16x32_bf16(false, ld_fragA(sW3, 1, lane), false, d1, (short)0, a3, false, false);
    if (lane < 16) {
      float nv = e31v + a3[0] + b3[0];
      e31[e] = nv;
      atomicAdd(&agg31[d], nv);
    }
  }
}

// ---------- node processor: [x|agg](64) -> 64 -> 64 -> 1 ----------
__global__ __launch_bounds__(128, 1) void node_kernel(
    const float* __restrict__ x, const float* __restrict__ aggc,
    const float* __restrict__ agg31, float* __restrict__ x31,
    const u16* __restrict__ W1s, const float* __restrict__ b1,
    const u16* __restrict__ W2s, const float* __restrict__ b2,
    const u16* __restrict__ W3s, const float* __restrict__ b3) {
  __shared__ __align__(16) u16 sW1[64 * 64];
  __shared__ __align__(16) u16 sW2[64 * 64];
  __shared__ __align__(16) u16 sW3[16 * 64];
  __shared__ __align__(16) u16 sH1[4][16 * HS];
  __shared__ __align__(16) u16 sH2[4][16 * HS];

  const int tid = threadIdx.x;
  const int wave = tid >> 5, lane = tid & 31;
  const int m = lane & 15, half = lane >> 4;

  {
    uint4* s1 = (uint4*)sW1; const uint4* g1 = (const uint4*)W1s;
    for (int i = tid; i < (64 * 64) / 8; i += 128) s1[i] = g1[i];
    uint4* s2 = (uint4*)sW2; const uint4* g2 = (const uint4*)W2s;
    for (int i = tid; i < (64 * 64) / 8; i += 128) s2[i] = g2[i];
    uint4* s3 = (uint4*)sW3; const uint4* g3 = (const uint4*)W3s;
    for (int i = tid; i < (16 * 64) / 8; i += 128) s3[i] = g3[i];
  }

  const int nbase = blockIdx.x * 64 + wave * 16;
  int node = nbase + m;
  const bool valid = node < NN;
  if (!valid) node = NN - 1;                 // clamp loads; store guarded below
  const float x31v = x31[node];

  v16bf bx = mk_fragB_g(x + (size_t)node * 32, half, x31v);
  v16bf bg = mk_fragB_g(aggc + (size_t)node * 32, half, agg31[node]);

  u16* h1 = sH1[wave];
  u16* h2 = sH2[wave];
  __syncthreads();

#pragma unroll
  for (int mt = 0; mt < 4; ++mt) {
    v8f acc = bias_acc(b1, mt, half);
    acc = __builtin_amdgcn_wmma_f32_16x16x32_bf16(false, ld_fragA(sW1, mt * 2 + 0, lane), false, bx, (short)0, acc, false, false);
    acc = __builtin_amdgcn_wmma_f32_16x16x32_bf16(false, ld_fragA(sW1, mt * 2 + 1, lane), false, bg, (short)0, acc, false, false);
    silu_store(h1 + m * HS + mt * 16 + half * 8, acc);
  }
  __syncthreads();

  {
    v16bf c0 = ld_fragB(h1 + m * HS, 0, half);
    v16bf c1 = ld_fragB(h1 + m * HS, 32, half);
#pragma unroll
    for (int mt = 0; mt < 4; ++mt) {
      v8f acc = bias_acc(b2, mt, half);
      acc = __builtin_amdgcn_wmma_f32_16x16x32_bf16(false, ld_fragA(sW2, mt * 2 + 0, lane), false, c0, (short)0, acc, false, false);
      acc = __builtin_amdgcn_wmma_f32_16x16x32_bf16(false, ld_fragA(sW2, mt * 2 + 1, lane), false, c1, (short)0, acc, false, false);
      silu_store(h2 + m * HS + mt * 16 + half * 8, acc);
    }
  }
  __syncthreads();

  {
    v16bf d0 = ld_fragB(h2 + m * HS, 0, half);
    v16bf d1 = ld_fragB(h2 + m * HS, 32, half);
    v8f a3 = {};
    a3 = __builtin_amdgcn_wmma_f32_16x16x32_bf16(false, ld_fragA(sW3, 0, lane), false, d0, (short)0, a3, false, false);
    a3 = __builtin_amdgcn_wmma_f32_16x16x32_bf16(false, ld_fragA(sW3, 1, lane), false, d1, (short)0, a3, false, false);
    if (lane < 16 && valid) x31[node] = x31v + a3[0] + b3[0];
  }
}

// ---------- final assembly into d_out = [x (N*32), edge_attr (E*32)] ----------
__global__ void k_out(const float* __restrict__ x, const float* __restrict__ ea,
                      const float* __restrict__ x31, const float* __restrict__ e31,
                      float* __restrict__ out) {
  long long i = (long long)blockIdx.x * blockDim.x + threadIdx.x;
  const long long nx = (long long)NN * 32;
  const long long ne = (long long)NE * 32;
  if (i < nx) {
    int c = (int)(i & 31);
    out[i] = (c == 31) ? x31[i >> 5] : x[i];
  } else if (i < nx + ne) {
    long long j = i - nx;
    int c = (int)(j & 31);
    out[i] = (c == 31) ? e31[j >> 5] : ea[j];
  }
}

static inline size_t alup(size_t v) { return (v + 255) & ~(size_t)255; }

extern "C" void kernel_launch(void* const* d_in, const int* in_sizes, int n_in,
                              void* d_out, int out_size, void* d_ws, size_t ws_size,
                              hipStream_t stream) {
  (void)in_sizes; (void)n_in; (void)out_size; (void)ws_size;
  const float* x    = (const float*)d_in[0];
  const float* ea   = (const float*)d_in[1];
  const int*   ei   = (const int*)d_in[2];
  const float* eW1f = (const float*)d_in[3];
  const float* eb1  = (const float*)d_in[4];
  const float* eW2f = (const float*)d_in[5];
  const float* eb2  = (const float*)d_in[6];
  const float* eW3  = (const float*)d_in[7];
  const float* eb3  = (const float*)d_in[8];
  const float* nW1f = (const float*)d_in[9];
  const float* nb1  = (const float*)d_in[10];
  const float* nW2f = (const float*)d_in[11];
  const float* nb2  = (const float*)d_in[12];
  const float* nW3  = (const float*)d_in[13];
  const float* nb3  = (const float*)d_in[14];
  const int* src = ei;
  const int* dst = ei + NE;

  // workspace layout
  char* w = (char*)d_ws;
  float* x31   = (float*)w;  w += alup((size_t)NN * 4);
  float* e31   = (float*)w;  w += alup((size_t)NE * 4);
  float* agg31 = (float*)w;  w += alup((size_t)NN * 4);
  float* aggc  = (float*)w;  w += alup((size_t)NN * 32 * 4);
  u16* eW1s = (u16*)w; w += alup((size_t)96 * 64 * 2);
  u16* eW2s = (u16*)w; w += alup((size_t)64 * 64 * 2);
  u16* eW3s = (u16*)w; w += alup((size_t)16 * 64 * 2);
  u16* nW1s = (u16*)w; w += alup((size_t)64 * 64 * 2);
  u16* nW2s = (u16*)w; w += alup((size_t)64 * 64 * 2);
  u16* nW3s = (u16*)w; w += alup((size_t)16 * 64 * 2);

  const int T = 256;
  // fragment-order bf16 weight swizzles (once per launch)
  k_swz<<<(96 * 64 + T - 1) / T, T, 0, stream>>>(eW1f, eW1s, 96, 64);
  k_swz<<<(64 * 64 + T - 1) / T, T, 0, stream>>>(eW2f, eW2s, 64, 64);
  k_swz<<<(64 * 64 + T - 1) / T, T, 0, stream>>>(nW1f, nW1s, 64, 64);
  k_swz<<<(64 * 64 + T - 1) / T, T, 0, stream>>>(nW2f, nW2s, 64, 64);
  k_swz_vec<<<(16 * 64 + T - 1) / T, T, 0, stream>>>(eW3, eW3s, 64);
  k_swz_vec<<<(16 * 64 + T - 1) / T, T, 0, stream>>>(nW3, nW3s, 64);
  // evolving scalar channels
  k_take31<<<(NN + T - 1) / T, T, 0, stream>>>(x, x31, NN);
  k_take31<<<(NE + T - 1) / T, T, 0, stream>>>(ea, e31, NE);
  // invariant part of the scatter-sum (channels 0..30), computed once
  k_zero<<<((NN * 32) + T - 1) / T, T, 0, stream>>>(aggc, NN * 32);
  k_aggconst<<<(NE + T - 1) / T, T, 0, stream>>>(ea, dst, aggc);

  for (int pass = 0; pass < 5; ++pass) {
    k_zero<<<(NN + T - 1) / T, T, 0, stream>>>(agg31, NN);
    edge_kernel<<<NE / 64, 128, 0, stream>>>(x, ea, src, dst, x31, e31, agg31,
                                             eW1s, eb1, eW2s, eb2, eW3s, eb3);
    node_kernel<<<(NN + 63) / 64, 128, 0, stream>>>(x, aggc, agg31, x31,
                                                    nW1s, nb1, nW2s, nb2, nW3s, nb3);
  }

  long long total = (long long)(NN + NE) * 32;
  k_out<<<(unsigned)((total + T - 1) / T), T, 0, stream>>>(x, ea, x31, e31, (float*)d_out);
}
